// MultiHeadAttention_15324443312188
// MI455X (gfx1250) — matrix operand order repaired and checked by reading, not compiled, not run on hardware
//
#include <hip/hip_runtime.h>
#include <hip/hip_bf16.h>

typedef __attribute__((ext_vector_type(16))) __bf16 v16bf;
typedef __attribute__((ext_vector_type(8)))  __bf16 v8bf;
typedef __attribute__((ext_vector_type(8)))  float  v8f;

#define D_MODEL 1024
#define N_HEAD  16
#define HEAD_DIM 64
#define SEQ     2048
#define BATCH   4
#define MTOT    (BATCH*SEQ)   // 8192

static __device__ __forceinline__ v16bf cat8(v8bf lo, v8bf hi) {
  return __builtin_shufflevector(lo, hi, 0,1,2,3,4,5,6,7,8,9,10,11,12,13,14,15);
}
static __device__ __forceinline__ v8f wmma_bf16(v16bf a, v16bf b, v8f c) {
  return __builtin_amdgcn_wmma_f32_16x16x32_bf16(false, a, false, b, (short)0, c,
                                                 false, false);
}

// ---------------- conversion kernels ----------------
__global__ void cvt_f32_to_bf16(const float* __restrict__ in,
                                __bf16* __restrict__ out, int n) {
  int i = blockIdx.x * blockDim.x + threadIdx.x;
  if (i < n) out[i] = (__bf16)in[i];
}

// out[n*K + k] = in[k*N + n]   (weight transpose so B-frags read contiguous K)
__global__ void cvt_transpose_bf16(const float* __restrict__ in,
                                   __bf16* __restrict__ out, int K, int N) {
  int i = blockIdx.x * blockDim.x + threadIdx.x;
  if (i < K * N) {
    int k = i / N, n = i % N;
    out[(size_t)n * K + k] = (__bf16)in[i];
  }
}

// ---------------- bf16 WMMA GEMM: C = A[M,K] * Wt[N,K]^T + bias ----------------
// mode 0: bf16 out [B,H,T,Dh]   (Q,K layout)
// mode 1: bf16 out [B,H,Dh,T]   (V transposed layout)
// mode 2: f32  out [M,N]        (final projection)
__global__ __launch_bounds__(128)
void gemm_bf16_wmma(const __bf16* __restrict__ A, const __bf16* __restrict__ Wt,
                    const float* __restrict__ bias, void* __restrict__ out,
                    int mode, int M, int N, int K) {
  const int lane  = threadIdx.x & 31;
  const int wave  = threadIdx.x >> 5;
  const int row16 = lane & 15;
  const int half  = lane >> 4;
  const int mBase = blockIdx.y * 128 + (wave >> 1) * 64;
  const int nBase = blockIdx.x * 128 + (wave & 1) * 64;

  const __bf16* aRow[4];
  const __bf16* bRow[4];
#pragma unroll
  for (int i = 0; i < 4; ++i) {
    aRow[i] = A  + (size_t)(mBase + i * 16 + row16) * K;
    bRow[i] = Wt + (size_t)(nBase + i * 16 + row16) * K;
  }

  v8f acc[4][4] = {};

  for (int k0 = 0; k0 < K; k0 += 32) {
    v16bf af[4], bf_[4];
#pragma unroll
    for (int i = 0; i < 4; ++i) {
      // A-fragment: lane row = lane%16, element e -> K = half*16 + e (contiguous)
      af[i] = *(const v16bf*)(aRow[i] + k0 + half * 16);
      // B-fragment: lane col = lane%16, element e -> K = half*16 + e (contiguous)
      bf_[i] = *(const v16bf*)(bRow[i] + k0 + half * 16);
    }
#pragma unroll
    for (int mi = 0; mi < 4; ++mi)
#pragma unroll
      for (int nj = 0; nj < 4; ++nj)
        acc[mi][nj] = wmma_bf16(af[mi], bf_[nj], acc[mi][nj]);
  }

  // epilogue: D layout -> M = r + 8*half, N = lane%16
#pragma unroll
  for (int mi = 0; mi < 4; ++mi) {
#pragma unroll
    for (int nj = 0; nj < 4; ++nj) {
      const int gN = nBase + nj * 16 + row16;
      const float bv = bias[gN];
#pragma unroll
      for (int r = 0; r < 8; ++r) {
        const int gM = mBase + mi * 16 + r + 8 * half;
        const float v = acc[mi][nj][r] + bv;
        if (mode == 2) {
          ((float*)out)[(size_t)gM * N + gN] = v;
        } else {
          const int b = gM >> 11, t = gM & (SEQ - 1);
          const int h = gN >> 6,  d = gN & (HEAD_DIM - 1);
          if (mode == 0)
            ((__bf16*)out)[(((size_t)(b * N_HEAD + h) * SEQ) + t) * HEAD_DIM + d] =
                (__bf16)v;
          else
            ((__bf16*)out)[(((size_t)(b * N_HEAD + h) * HEAD_DIM) + d) * SEQ + t] =
                (__bf16)v;
        }
      }
    }
  }
}

// ---------------- causal flash attention ----------------
// Q,K: [B,H,T,Dh] bf16 ; Vt: [B,H,Dh,T] bf16 ; Ob: [B*T, D_MODEL] bf16
// One wave handles 16 query rows. Computes S^T = K*Q^T (softmax over keys is
// in-lane + one shfl_xor(16)), then O^T = Vt * P^T with P staged through LDS.
__global__ __launch_bounds__(128)
void flash_attn_wmma(const __bf16* __restrict__ Q, const __bf16* __restrict__ Kb,
                     const __bf16* __restrict__ Vt, __bf16* __restrict__ Ob) {
  __shared__ __bf16 pls[4 * 16 * 32];  // per-wave [16 q][32 key] bf16 tile

  const int lane = threadIdx.x & 31;
  const int wave = threadIdx.x >> 5;
  const int qcol = lane & 15;
  const int half = lane >> 4;

  const int bh = blockIdx.x >> 5;                       // B*H = 64
  const int q0 = (blockIdx.x & 31) * 64 + wave * 16;    // 32 q-blocks of 64

  const __bf16* Kbh = Kb + (size_t)bh * SEQ * HEAD_DIM;
  const __bf16* Vbh = Vt + (size_t)bh * HEAD_DIM * SEQ;

  // Q as B-fragment of S^T = K * Q^T : lane col = q, elements -> d contiguous
  const __bf16* qrow = Q + ((size_t)bh * SEQ + q0 + qcol) * HEAD_DIM;
  const v16bf bq0 = *(const v16bf*)(qrow + half * 16);
  const v16bf bq1 = *(const v16bf*)(qrow + 32 + half * 16);

  v8f o[4] = {};
  float m = -1e30f, l = 0.f;
  __bf16* myrow = pls + wave * 512 + qcol * 32;

  const int qg = q0 + qcol;

  for (int k0 = 0; k0 < q0 + 16; k0 += 32) {
    v8f st[2];
#pragma unroll
    for (int s = 0; s < 2; ++s) {
      const int kbase = k0 + s * 16;
      if (kbase < q0 + 16) {
        const __bf16* krow = Kbh + (size_t)(kbase + qcol) * HEAD_DIM;
        const v16bf ak0 = *(const v16bf*)(krow + half * 16);
        const v16bf ak1 = *(const v16bf*)(krow + 32 + half * 16);
        v8f s8 = {};
        s8 = wmma_bf16(ak0, bq0, s8);
        s8 = wmma_bf16(ak1, bq1, s8);
#pragma unroll
        for (int r = 0; r < 8; ++r) {
          const int kg = kbase + r + 8 * half;           // M = key index
          const float v = s8[r] * 0.125f;                // 1/sqrt(64)
          st[s][r] = (kg > qg) ? -1e30f : v;             // causal mask
        }
      } else {
#pragma unroll
        for (int r = 0; r < 8; ++r) st[s][r] = -1e30f;
      }
    }

    // online softmax over this 32-key block (per q = lane%16)
    float mx = st[0][0];
#pragma unroll
    for (int r = 1; r < 8; ++r) mx = fmaxf(mx, st[0][r]);
#pragma unroll
    for (int r = 0; r < 8; ++r) mx = fmaxf(mx, st[1][r]);
    mx = fmaxf(mx, __shfl_xor(mx, 16));
    const float mnew = fmaxf(m, mx);
    const float scale = __expf(m - mnew);

    float psum = 0.f;
#pragma unroll
    for (int s = 0; s < 2; ++s)
#pragma unroll
      for (int r = 0; r < 8; ++r) {
        const float p = __expf(st[s][r] - mnew);
        psum += p;
        myrow[s * 16 + r + 8 * half] = (__bf16)p;   // P^T staged as [q][key]
      }
    psum += __shfl_xor(psum, 16);
    l = l * scale + psum;
    m = mnew;
#pragma unroll
    for (int dt = 0; dt < 4; ++dt) o[dt] = o[dt] * scale;

    // order the wave-private LDS store -> B-fragment load
    asm volatile("s_wait_dscnt 0" ::: "memory");

    const v16bf pb = *(const v16bf*)(pls + wave * 512 + qcol * 32 + half * 16);
#pragma unroll
    for (int dt = 0; dt < 4; ++dt) {
      const __bf16* vrow = Vbh + (size_t)(dt * 16 + qcol) * SEQ + k0;
      const v16bf av = *(const v16bf*)(vrow + half * 16);
      o[dt] = wmma_bf16(av, pb, o[dt]);
    }
  }

  const float inv = 1.0f / l;
  const int b = bh >> 4, h = bh & (N_HEAD - 1);
#pragma unroll
  for (int dt = 0; dt < 4; ++dt)
#pragma unroll
    for (int r = 0; r < 8; ++r) {
      const int d = dt * 16 + r + 8 * half;
      Ob[((size_t)(b * SEQ + qg)) * D_MODEL + h * HEAD_DIM + d] =
          (__bf16)(o[dt][r] * inv);
    }
}

// ---------------- launch ----------------
extern "C" void kernel_launch(void* const* d_in, const int* in_sizes, int n_in,
                              void* d_out, int out_size, void* d_ws, size_t ws_size,
                              hipStream_t stream) {
  const float* x  = (const float*)d_in[0];
  const float* Wq = (const float*)d_in[1];
  const float* bq = (const float*)d_in[2];
  const float* Wk = (const float*)d_in[3];
  const float* bk = (const float*)d_in[4];
  const float* Wv = (const float*)d_in[5];
  const float* bv = (const float*)d_in[6];
  const float* Wo = (const float*)d_in[7];
  const float* bo = (const float*)d_in[8];

  char* ws = (char*)d_ws;
  const size_t XB = (size_t)MTOT * D_MODEL * sizeof(__bf16);       // 16 MB
  const size_t WB = (size_t)D_MODEL * D_MODEL * sizeof(__bf16);    // 2 MB
  __bf16* xbf = (__bf16*)(ws);             // x bf16; later reused as attn output
  __bf16* wqT = (__bf16*)(ws + XB);
  __bf16* wkT = (__bf16*)(ws + XB + WB);
  __bf16* wvT = (__bf16*)(ws + XB + 2 * WB);
  __bf16* woT = (__bf16*)(ws + XB + 3 * WB);
  __bf16* Qb  = (__bf16*)(ws + XB + 4 * WB);
  __bf16* Kb  = (__bf16*)(ws + XB + 4 * WB + XB);
  __bf16* Vtb = (__bf16*)(ws + XB + 4 * WB + 2 * XB);

  const int nX = MTOT * D_MODEL;           // 8388608
  const int nW = D_MODEL * D_MODEL;        // 1048576

  cvt_f32_to_bf16<<<nX / 256, 256, 0, stream>>>(x, xbf, nX);
  cvt_transpose_bf16<<<nW / 256, 256, 0, stream>>>(Wq, wqT, D_MODEL, D_MODEL);
  cvt_transpose_bf16<<<nW / 256, 256, 0, stream>>>(Wk, wkT, D_MODEL, D_MODEL);
  cvt_transpose_bf16<<<nW / 256, 256, 0, stream>>>(Wv, wvT, D_MODEL, D_MODEL);
  cvt_transpose_bf16<<<nW / 256, 256, 0, stream>>>(Wo, woT, D_MODEL, D_MODEL);

  dim3 ggrid(D_MODEL / 128, MTOT / 128);   // 8 x 64
  gemm_bf16_wmma<<<ggrid, 128, 0, stream>>>(xbf, wqT, bq, Qb, 0,
                                            MTOT, D_MODEL, D_MODEL);
  gemm_bf16_wmma<<<ggrid, 128, 0, stream>>>(xbf, wkT, bk, Kb, 0,
                                            MTOT, D_MODEL, D_MODEL);
  gemm_bf16_wmma<<<ggrid, 128, 0, stream>>>(xbf, wvT, bv, Vtb, 1,
                                            MTOT, D_MODEL, D_MODEL);

  // x bf16 no longer needed -> reuse as attention output buffer
  flash_attn_wmma<<<BATCH * N_HEAD * (SEQ / 64), 128, 0, stream>>>(Qb, Kb, Vtb, xbf);

  gemm_bf16_wmma<<<ggrid, 128, 0, stream>>>(xbf, woT, bo, d_out, 2,
                                            MTOT, D_MODEL, D_MODEL);
}
